// Attention_46291157516866
// MI455X (gfx1250) — compile-verified, matrix operand order checked by reading, not run on hardware
//
#include <hip/hip_runtime.h>
#include <hip/hip_bf16.h>
#include <math.h>

// ---------------- problem constants (match reference) ----------------
constexpr int Bsz = 4;          // sequences
constexpr int Sq  = 1024;       // tokens per sequence
constexpr int NT  = Bsz * Sq;   // total tokens = 4096
constexpr int NH  = 16;         // query heads
constexpr int NHk = 4;          // kv heads
constexpr int Gq  = NH / NHk;   // 4
constexpr int Dh  = 128;        // head dim
constexpr float SCALE = 0.08838834764831845f;

// ---------------- tiling ----------------
constexpr int BM = 128;         // q rows per workgroup (8 waves x 16 rows)
constexpr int BN = 32;          // keys per iteration (2 x 16-wide WMMA tiles)
constexpr int NWAVES = 8;

typedef __attribute__((ext_vector_type(8)))  float  v8f;
typedef __attribute__((ext_vector_type(16))) __bf16 v16bf;

static __device__ __forceinline__ __bf16 f2bf(float f) { return (__bf16)f; }

// A-matrix (and assumed-symmetric B-matrix) K index for 16-bit 16x32 WMMA
// fragments: vector element e = 2p+o lives in VGPR p; lanes 0-15 get K base 0,
// lanes 16-31 get K base +8 (hiK).
static __device__ __forceinline__ int kmap(int p, int o, int hiK) {
    return ((p < 4) ? (2 * p + o) : (16 + 2 * (p - 4) + o)) + hiK;
}

// ======================= flash attention forward =======================
__global__ __launch_bounds__(256)
void fa_fwd_kernel(const float* __restrict__ Qg, const float* __restrict__ Kg,
                   const float* __restrict__ Vg, float* __restrict__ Og)
{
    constexpr int DPAD = Dh + 4;    // sK row pitch  (132 bf16 = 66 words)
    constexpr int TPAD = BN + 2;    // sVt row pitch (34 bf16 = 17 words, conflict-free)
    constexpr int PST  = BN + 2;    // sP  row pitch
    __shared__ __bf16 sK [2][BN][DPAD];   // K block, [key][d]
    __shared__ __bf16 sVt[2][Dh][TPAD];   // V block TRANSPOSED, [d][key]
    __shared__ __bf16 sP [NWAVES][16][PST];

    const int qb  = blockIdx.x;        // q block within sequence
    const int h   = blockIdx.y;        // query head
    const int b   = blockIdx.z;        // sequence
    const int hk  = h / Gq;            // kv head (GQA)
    const int tid = threadIdx.x;
    const int wave = tid >> 5;
    const int lane = tid & 31;
    const int ln16 = lane & 15;
    const int hiK  = (lane >> 4) << 3; // 0 or 8 (K offset, A/B frags; M offset, C/D frags)
    const int qbase = qb * BM;
    const int wrow  = qbase + wave * 16;   // first in-seq q row for this wave

    // ---- load this wave's 16x128 Q strip as 4 bf16 A-fragments (scale folded in) ----
    v16bf aQ[4];
    {
        const size_t tok = (size_t)b * Sq + wrow + ln16;   // M = ln16
        const float* qrow = Qg + (tok * NH + h) * Dh;
#pragma unroll
        for (int f = 0; f < 4; ++f)
#pragma unroll
            for (int p = 0; p < 8; ++p)
#pragma unroll
                for (int o = 0; o < 2; ++o)
                    aQ[f][2 * p + o] = f2bf(qrow[32 * f + kmap(p, o, hiK)] * SCALE);
    }

    const v8f vzero = {0.f, 0.f, 0.f, 0.f, 0.f, 0.f, 0.f, 0.f};
    v8f accO[8];
#pragma unroll
    for (int t = 0; t < 8; ++t) accO[t] = vzero;
    float mrow[8], lrow[8];
#pragma unroll
    for (int r = 0; r < 8; ++r) { mrow[r] = -INFINITY; lrow[r] = 0.f; }

    // ---- double-buffered staging: each thread owns 4 float4 of K and V ----
    // flat float4 index i4 = tid + j*256 over BN*Dh/4 = 1024 slots
    // key row = i4>>5, d col = (i4&31)*4  (16B-aligned global loads)
    float4 kreg[4], vreg[4];
    auto prefetch = [&](int k0) {
#pragma unroll
        for (int j = 0; j < 4; ++j) {
            const int i4 = tid + j * 256;
            const int kr = i4 >> 5;
            const int dc = (i4 & 31) << 2;
            const size_t tok  = (size_t)b * Sq + k0 + kr;
            const size_t base = (tok * NHk + hk) * Dh + dc;
            kreg[j] = *(const float4*)(Kg + base);
            vreg[j] = *(const float4*)(Vg + base);
        }
    };

    const int nkb = (qbase + BM) / BN;       // causal: only blocks up to this q block
    int buf = 0;
    prefetch(0);
    for (int kb = 0; kb < nkb; ++kb) {
        // ---- store staged registers into LDS (fp32 -> bf16; V transposed) ----
#pragma unroll
        for (int j = 0; j < 4; ++j) {
            const int i4 = tid + j * 256;
            const int kr = i4 >> 5;
            const int dc = (i4 & 31) << 2;
            sK[buf][kr][dc + 0] = f2bf(kreg[j].x);
            sK[buf][kr][dc + 1] = f2bf(kreg[j].y);
            sK[buf][kr][dc + 2] = f2bf(kreg[j].z);
            sK[buf][kr][dc + 3] = f2bf(kreg[j].w);
            sVt[buf][dc + 0][kr] = f2bf(vreg[j].x);
            sVt[buf][dc + 1][kr] = f2bf(vreg[j].y);
            sVt[buf][dc + 2][kr] = f2bf(vreg[j].z);
            sVt[buf][dc + 3][kr] = f2bf(vreg[j].w);
        }
        if (kb + 1 < nkb) prefetch((kb + 1) * BN);   // overlaps with compute below
        __syncthreads();

        const int k0 = kb * BN;
        if (k0 <= wrow + 15) {   // wave-uniform: block not entirely above the diagonal
            // ---- scores: S = Q * K^T  (2 tiles of 16 keys, 4 K-chunks each) ----
            v8f accS[2];
#pragma unroll
            for (int t = 0; t < 2; ++t) {
                const __bf16* krow = &sK[buf][t * 16 + ln16][0];
                v16bf bK[4];
#pragma unroll
                for (int f = 0; f < 4; ++f)
#pragma unroll
                    for (int p = 0; p < 8; ++p)
#pragma unroll
                        for (int o = 0; o < 2; ++o)
                            bK[f][2 * p + o] = krow[32 * f + kmap(p, o, hiK)];
                v8f c = vzero;
#pragma unroll
                for (int f = 0; f < 4; ++f)
                    c = __builtin_amdgcn_wmma_f32_16x16x32_bf16(
                            false, aQ[f], false, bK[f], (short)0, c, false, false);
                accS[t] = c;
            }

            // ---- causal mask + online softmax (row = C-layout VGPR index) ----
            __bf16* sPw = &sP[wave][0][0];
#pragma unroll
            for (int r = 0; r < 8; ++r) {
                const int qrow = wrow + r + hiK;         // this lane's row for VGPR r
                float s0 = accS[0][r];
                float s1 = accS[1][r];
                if (k0 + ln16 > qrow)       s0 = -1e30f;
                if (k0 + 16 + ln16 > qrow)  s1 = -1e30f;
                float mb = fmaxf(s0, s1);
#pragma unroll
                for (int off = 8; off >= 1; off >>= 1)
                    mb = fmaxf(mb, __shfl_xor(mb, off, 32));
                const float mn    = fmaxf(mrow[r], mb);
                const float alpha = __expf(mrow[r] - mn);   // -inf first time -> 0
                const float p0 = __expf(s0 - mn);
                const float p1 = __expf(s1 - mn);
                float rs = p0 + p1;
#pragma unroll
                for (int off = 8; off >= 1; off >>= 1)
                    rs += __shfl_xor(rs, off, 32);
                lrow[r] = lrow[r] * alpha + rs;
                mrow[r] = mn;
#pragma unroll
                for (int t = 0; t < 8; ++t) accO[t][r] *= alpha;
                // stash P (C-layout -> wave-private LDS, row-major 16x32)
                sPw[(r + hiK) * PST + ln16]      = f2bf(p0);
                sPw[(r + hiK) * PST + 16 + ln16] = f2bf(p1);
            }
            // wave-local LDS RAW: drain DS ops before re-reading P
            asm volatile("s_wait_dscnt 0" ::: "memory");

            // ---- re-read P in A-layout (16 rows x 32 keys, K = key index) ----
            v16bf aP;
#pragma unroll
            for (int p = 0; p < 8; ++p)
#pragma unroll
                for (int o = 0; o < 2; ++o)
                    aP[2 * p + o] = sPw[ln16 * PST + kmap(p, o, hiK)];

            // ---- O += P * V : 8 output d-tiles, K-dim = 32 keys ----
            // sVt is [d][key] so the B-fragment K-walk is contiguous (merges to b64)
#pragma unroll
            for (int dt = 0; dt < 8; ++dt) {
                const __bf16* vrow = &sVt[buf][dt * 16 + ln16][0];
                v16bf bV;
#pragma unroll
                for (int p = 0; p < 8; ++p)
#pragma unroll
                    for (int o = 0; o < 2; ++o)
                        bV[2 * p + o] = vrow[kmap(p, o, hiK)];
                accO[dt] = __builtin_amdgcn_wmma_f32_16x16x32_bf16(
                               false, aP, false, bV, (short)0, accO[dt], false, false);
            }
        }
        buf ^= 1;
    }

    // ---- epilogue: normalize and store fp32 output ----
#pragma unroll
    for (int r = 0; r < 8; ++r) {
        const float inv = 1.0f / lrow[r];
        const size_t tok = (size_t)b * Sq + wrow + r + hiK;
        float* orow = Og + (tok * NH + h) * Dh;
#pragma unroll
        for (int dt = 0; dt < 8; ++dt)
            orow[dt * 16 + ln16] = accO[dt][r] * inv;
    }
}

// ======================= KV-cache copy + scatter (float4) =======================
__global__ void cache_copy_kernel(const float4* __restrict__ kc_in,
                                  const float4* __restrict__ vc_in,
                                  float4* __restrict__ kc_out,
                                  float4* __restrict__ vc_out, int n4)
{
    int i = blockIdx.x * blockDim.x + threadIdx.x;
    if (i < n4) { kc_out[i] = kc_in[i]; vc_out[i] = vc_in[i]; }
}

__global__ void cache_scatter_kernel(const float4* __restrict__ k,
                                     const float4* __restrict__ v,
                                     const int* __restrict__ slot,
                                     float4* __restrict__ kc_out,
                                     float4* __restrict__ vc_out, int ntok, int C4)
{
    int i = blockIdx.x * blockDim.x + threadIdx.x;
    if (i >= ntok * C4) return;
    const int t = i / C4, c = i - t * C4;
    const int s = slot[t];
    if (s >= 0) {
        kc_out[(size_t)s * C4 + c] = k[i];
        vc_out[(size_t)s * C4 + c] = v[i];
    }
}

// ======================= launch =======================
extern "C" void kernel_launch(void* const* d_in, const int* in_sizes, int n_in,
                              void* d_out, int out_size, void* d_ws, size_t ws_size,
                              hipStream_t stream)
{
    const float* q    = (const float*)d_in[0];
    const float* k    = (const float*)d_in[1];
    const float* v    = (const float*)d_in[2];
    const float* kc   = (const float*)d_in[3];
    const float* vc   = (const float*)d_in[4];
    const int*   slot = (const int*)d_in[5];

    float* o_out  = (float*)d_out;                       // [NT, NH, Dh]
    float* kc_out = o_out + (size_t)NT * NH * Dh;        // [NT, NHk*Dh]
    float* vc_out = kc_out + (size_t)NT * NHk * Dh;

    const int C4     = (NHk * Dh) / 4;     // 128 float4 per token
    const int cache4 = NT * C4;            // 524,288 float4

    cache_copy_kernel<<<(cache4 + 255) / 256, 256, 0, stream>>>(
        (const float4*)kc, (const float4*)vc, (float4*)kc_out, (float4*)vc_out, cache4);
    cache_scatter_kernel<<<(cache4 + 255) / 256, 256, 0, stream>>>(
        (const float4*)k, (const float4*)v, slot, (float4*)kc_out, (float4*)vc_out, NT, C4);

    dim3 grid(Sq / BM, NH, Bsz);       // (8, 16, 4)
    fa_fwd_kernel<<<grid, 256, 0, stream>>>(q, k, v, o_out);
}
